// NeRFSmall_90220083020381
// MI455X (gfx1250) — matrix operand order repaired and checked by reading, not compiled, not run on hardware
//
#include <hip/hip_runtime.h>
#include <hip/hip_bf16.h>
#include <math.h>

typedef __attribute__((ext_vector_type(16))) _Float16 v16h;
typedef __attribute__((ext_vector_type(8)))  _Float16 v8h;
typedef __attribute__((ext_vector_type(8)))  float    v8f;
typedef __attribute__((ext_vector_type(4)))  unsigned int u32x4;
typedef __attribute__((ext_vector_type(8)))  int         i32x8;
typedef __attribute__((ext_vector_type(4)))  int         i32x4;

#define HASH_T      (1u << 19)
#define HASH_MASK   (HASH_T - 1u)
#define HASH_P1     2654435761u

// Swizzled f16 weight blob offsets (in halfs). Layout: per (kb,nb) 32x16 block,
// 512 halfs: lane l owns halfs [l*16 .. l*16+15], element e -> B[kb*32 + (l>>4)*16 + e][nb*16 + (l&15)].
#define W0_OFF 0      // dW0: 40x64  -> 64x64  (2 kb x 4 nb)
#define W1_OFF 4096   // dW1: 64x65  -> 64x80  (2 kb x 5 nb)
#define S0_OFF 9216   // sW0: 104x64 -> 128x64 (4 kb x 4 nb)
#define S1_OFF 17408  // sW1: 64x64  -> 64x64  (2 kb x 4 nb)
#define S2_OFF 21504  // sW2: 64x39  -> 64x48  (2 kb x 3 nb)
#define WGT_TOTAL 24576                  // halfs = 49152 bytes = 12288 dwords

__device__ __forceinline__ uint32_t hash2(uint32_t ix, uint32_t iy) {
    return (ix ^ (iy * HASH_P1)) & HASH_MASK;
}

// ---- WMMA fragment helpers -------------------------------------------------

// A fragment (16x32 f16, MxK): lane l holds row m = l&15.
// elements 0..7  = K (kb*32 + (l>>4)*16 + 0..7)        -> one b128 LDS load
// elements 8..15 = K (kb*32 + (l>>4)*16 + 8..15)       -> one b128 LDS load
template<int KB>
__device__ __forceinline__ void load_a_frags(const _Float16* tile, int stride, v16h* a) {
    const int lane = threadIdx.x & 31;
    const int m = lane & 15;
    const int kh = (lane >> 4) & 1;
#pragma unroll
    for (int kb = 0; kb < KB; ++kb) {
        const _Float16* p = tile + m * stride + kb * 32 + kh * 16;
        v8h lo = *(const v8h*)(p);
        v8h hi = *(const v8h*)(p + 8);
        v16h r;
#pragma unroll
        for (int i = 0; i < 8; ++i) { r[i] = lo[i]; r[i + 8] = hi[i]; }
        a[kb] = r;
    }
}

// B fragment from TDM-staged LDS weights: 16 contiguous halfs (32B) per lane.
template<int KB>
__device__ __forceinline__ v8f gemm_acc(const v16h* a, const _Float16* wbase, int NB, int nb) {
    const int lane = threadIdx.x & 31;
    v8f acc = {};
#pragma unroll
    for (int kb = 0; kb < KB; ++kb) {
        v16h bf = *(const v16h*)(wbase + (((kb * NB + nb) << 9) + lane * 16));
        acc = __builtin_amdgcn_wmma_f32_16x16x32_f16(
            /*neg_a=*/false, a[kb], /*neg_b=*/false, bf,
            /*c_mod=*/(short)0, acc, /*reuse_a=*/false, /*reuse_b=*/false);
    }
    return acc;
}

// D layout: VGPR j, lane l -> row m = j + 8*(l>>4), col n = (l&15).
__device__ __forceinline__ void store_relu_tile(_Float16* dst, int stride, const float* bias,
                                                int nb, const v8f& acc) {
    const int lane = threadIdx.x & 31;
    const int n = (lane & 15) + nb * 16;
    const int mh = (lane >> 4) * 8;
    const float b = bias[n];
#pragma unroll
    for (int j = 0; j < 8; ++j) {
        float v = acc[j] + b;
        v = v > 0.f ? v : 0.f;
        dst[(mh + j) * stride + n] = (_Float16)v;
    }
}

// ---- Kernel 1: weight prep (f32 -> f16 fragment-swizzled, zero-padded) -----

__global__ __launch_bounds__(256) void prep_weights(const float* dW0, const float* dW1,
                                                    const float* sW0, const float* sW1,
                                                    const float* sW2, _Float16* dst) {
    int idx = blockIdx.x * blockDim.x + threadIdx.x;
    if (idx >= WGT_TOTAL) return;
    const float* src; int Ks, Ns, Np, off;
    if      (idx < W1_OFF) { src = dW0; Ks = 40;  Ns = 64; Np = 64; off = W0_OFF; }
    else if (idx < S0_OFF) { src = dW1; Ks = 64;  Ns = 65; Np = 80; off = W1_OFF; }
    else if (idx < S1_OFF) { src = sW0; Ks = 104; Ns = 64; Np = 64; off = S0_OFF; }
    else if (idx < S2_OFF) { src = sW1; Ks = 64;  Ns = 64; Np = 64; off = S1_OFF; }
    else                   { src = sW2; Ks = 64;  Ns = 39; Np = 48; off = S2_OFF; }
    int rel = idx - off;
    int NB = Np >> 4;
    int blk = rel >> 9, within = rel & 511;
    int lane = within >> 4, e = within & 15;
    int kb = blk / NB, nb = blk % NB;
    int k = kb * 32 + ((lane >> 4) & 1) * 16 + e;
    int n = nb * 16 + (lane & 15);
    float v = (k < Ks && n < Ns) ? src[k * Ns + n] : 0.f;
    dst[idx] = (_Float16)v;
}

// ---- Kernel 2: fused hash-encode + depth MLP + shading MLP -----------------
// 128 threads = 4 waves; each wave owns a 16-pixel row tile.
// Weights are DMA'd once per workgroup into LDS by the Tensor Data Mover,
// overlapped with the hash-encoding gather phase.

__global__ __launch_bounds__(128) void fused_mlp(
    const float* __restrict__ nxy, const float* __restrict__ sxy,
    const float* __restrict__ mv,  const float* __restrict__ tab,
    const _Float16* __restrict__ wgt,
    const float* __restrict__ db0, const float* __restrict__ db1,
    const float* __restrict__ sb0, const float* __restrict__ sb1,
    const float* __restrict__ sb2,
    float* __restrict__ out, int P) {

    __shared__ _Float16 sIn[4][16 * 128]; // per-wave MLP2 input: [0:2)nxy [2:34)hash [34:40)mv [40:104)feat [104:128)=0
    __shared__ _Float16 sT [4][16 * 64];  // per-wave hidden activations
    __shared__ __align__(64) _Float16 sWgt[WGT_TOTAL]; // TDM-staged swizzled weights (48KB)

    const int wv   = threadIdx.x >> 5;
    const int lane = threadIdx.x & 31;
    const int m    = lane & 15;
    const int sub  = lane >> 4;            // 2 lanes per pixel for encoding
    const int tileBase = blockIdx.x * 64 + wv * 16;
    const int pix  = tileBase + m;

    _Float16* bi = sIn[wv];
    _Float16* bt = sT[wv];

#if defined(__has_builtin) && __has_builtin(__builtin_amdgcn_tensor_load_to_lds)
    // ---- TDM: DMA the 48KB weight blob global -> LDS (wave 0 only; EXEC ignored)
    if (wv == 0) {
        const uint64_t ga = (uint64_t)(uintptr_t)wgt;
        const uint32_t lds_off = (uint32_t)(uintptr_t)(void*)&sWgt[0];
        u32x4 g0;
        g0[0] = 1u;                                               // count=1, user D#
        g0[1] = lds_off;                                          // lds_addr
        g0[2] = (uint32_t)ga;                                     // global_addr[31:0]
        g0[3] = (uint32_t)((ga >> 32) & 0x01FFFFFFu) | 0x80000000u; // addr[56:32] | type=2
        i32x8 g1;
        g1[0] = (int)(2u << 16);          // workgroup_mask=0, data_size=2 (4B)
        g1[1] = (int)(3072u << 16);       // tensor_dim0[15:0]=3072 (dwords per row)
        g1[2] = (int)(4u << 16);          // tensor_dim0[31:16]=0, tensor_dim1[15:0]=4
        g1[3] = (int)(3072u << 16);       // tensor_dim1[31:16]=0, tile_dim0=3072
        g1[4] = 4;                        // tile_dim1=4, tile_dim2=0
        g1[5] = 3072;                     // tensor_dim0_stride[31:0]
        g1[6] = 0;                        // stride hi / tensor_dim1_stride lo
        g1[7] = 0;
        i32x4 g2 = {0, 0, 0, 0};
        i32x4 g3 = {0, 0, 0, 0};
        i32x8 g4 = {0, 0, 0, 0, 0, 0, 0, 0};
        __builtin_amdgcn_tensor_load_to_lds(g0, g1, g2, g3, g4, 0);
    }
#else
    // Fallback: cooperative vector copy global -> LDS
    for (int i = threadIdx.x; i < WGT_TOTAL / 8; i += 128)
        ((v8h*)sWgt)[i] = ((const v8h*)wgt)[i];
#endif

    // ---- encoding phase (overlaps the TDM transfer) ----
    for (int i = 40 + sub; i < 128; i += 2) bi[m * 128 + i] = (_Float16)0.f;
    if (sub == 0) {
        bi[m * 128 + 0] = (_Float16)nxy[pix * 2 + 0];
        bi[m * 128 + 1] = (_Float16)nxy[pix * 2 + 1];
#pragma unroll
        for (int i = 0; i < 6; ++i) bi[m * 128 + 34 + i] = (_Float16)mv[pix * 6 + i];
    }
    {
        const float sx = sxy[pix * 2 + 0];
        const float sy = sxy[pix * 2 + 1];
        const float2* t2 = (const float2*)tab;
#pragma unroll
        for (int t = 0; t < 8; ++t) {
            const int lvl = sub * 8 + t;
            // res_l = floor(16 * b^l), b chosen so res_15 = 384; log2(b) = (log2(384)-4)/15
            const float res = floorf(exp2f(4.0f + (float)lvl * 0.30566416671474374f) + 1e-3f);
            const float px = sx * res, py = sy * res;
            const float fx = floorf(px), fy = floorf(py);
            const uint32_t x0 = (uint32_t)(int)fx, y0 = (uint32_t)(int)fy;
            const float wx = px - fx, wy = py - fy;
            const float2* tl = t2 + (size_t)lvl * HASH_T;
            const float2 c00 = tl[hash2(x0,     y0)];
            const float2 c10 = tl[hash2(x0 + 1, y0)];
            const float2 c01 = tl[hash2(x0,     y0 + 1)];
            const float2 c11 = tl[hash2(x0 + 1, y0 + 1)];
            const float w00 = (1.f - wx) * (1.f - wy), w10 = wx * (1.f - wy);
            const float w01 = (1.f - wx) * wy,         w11 = wx * wy;
            const float f0 = c00.x * w00 + c10.x * w10 + c01.x * w01 + c11.x * w11;
            const float f1 = c00.y * w00 + c10.y * w10 + c01.y * w01 + c11.y * w11;
            bi[m * 128 + 2 + lvl * 2 + 0] = (_Float16)f0;
            bi[m * 128 + 2 + lvl * 2 + 1] = (_Float16)f1;
        }
    }

#if defined(__has_builtin) && __has_builtin(__builtin_amdgcn_tensor_load_to_lds)
    if (wv == 0) __builtin_amdgcn_s_wait_tensorcnt(0);
#endif
    __syncthreads();

    v16h a[4];

    // ---- depth MLP layer 0: enc(40->64pad) @ dW0 -> relu -> bt[16][64] ----
    load_a_frags<2>(bi, 128, a);
#pragma unroll
    for (int nb = 0; nb < 4; ++nb) {
        v8f acc = gemm_acc<2>(a, sWgt + W0_OFF, 4, nb);
        store_relu_tile(bt, 64, db0, nb, acc);
    }
    __syncthreads();

    // ---- depth MLP layer 1: 64 @ dW1(64x65->80pad): col0 -> depth, 1..64 -> feat ----
    load_a_frags<2>(bt, 64, a);
#pragma unroll
    for (int nb = 0; nb < 5; ++nb) {
        v8f acc = gemm_acc<2>(a, sWgt + W1_OFF, 5, nb);
        const int cn = nb * 16 + (lane & 15);
        const float b = (cn < 65) ? db1[cn] : 0.f;
#pragma unroll
        for (int j = 0; j < 8; ++j) {
            const float v = acc[j] + b;
            const int mm = j + sub * 8;
            const int pp = tileBase + mm;
            if (cn == 0)      out[pp] = fabsf(v) + 1e-6f;                     // depth
            else if (cn < 65) bi[mm * 128 + 40 + (cn - 1)] = (_Float16)fmaxf(v, 0.f); // feat
        }
    }
    __syncthreads();

    // ---- shading MLP layer 0: [enc,feat](104->128pad) @ sW0 -> relu ----
    load_a_frags<4>(bi, 128, a);
#pragma unroll
    for (int nb = 0; nb < 4; ++nb) {
        v8f acc = gemm_acc<4>(a, sWgt + S0_OFF, 4, nb);
        store_relu_tile(bt, 64, sb0, nb, acc);
    }
    __syncthreads();

    // ---- shading MLP layer 1: 64 @ sW1 -> relu (in-place; A preloaded) ----
    load_a_frags<2>(bt, 64, a);
#pragma unroll
    for (int nb = 0; nb < 4; ++nb) {
        v8f acc = gemm_acc<2>(a, sWgt + S1_OFF, 4, nb);
        store_relu_tile(bt, 64, sb1, nb, acc);
    }
    __syncthreads();

    // ---- shading MLP layer 2: 64 @ sW2(64x39->48pad): diff / spec ----
    load_a_frags<2>(bt, 64, a);
#pragma unroll
    for (int nb = 0; nb < 3; ++nb) {
        v8f acc = gemm_acc<2>(a, sWgt + S2_OFF, 3, nb);
        const int cn = nb * 16 + (lane & 15);
        const float b = (cn < 39) ? sb2[cn] : 0.f;
#pragma unroll
        for (int j = 0; j < 8; ++j) {
            const float v = acc[j] + b;
            const int mm = j + sub * 8;
            const size_t pp = (size_t)(tileBase + mm);
            if (cn < 3)       out[(size_t)4 * P + pp * 3 + cn] = fabsf(v);    // diff
            else if (cn < 39) out[(size_t)7 * P + pp * 36 + (cn - 3)] = v;    // spec
        }
    }
}

// ---- Kernel 3: PCA surface normal from depth -------------------------------
// smallest right-singular vector of the 9x3 centered-neighborhood matrix
// == eigenvector of the 3x3 covariance with smallest eigenvalue (closed form).

__global__ __launch_bounds__(256) void normal_kernel(const float* __restrict__ invK_T,
                                                     const float* __restrict__ depth,
                                                     float* __restrict__ out,
                                                     int P, int W, int H) {
    const int pix = blockIdx.x * blockDim.x + threadIdx.x;
    if (pix >= P) return;
    const int r = pix / W, c = pix % W;
    const float K00 = invK_T[0], K01 = invK_T[1], K02 = invK_T[2];
    const float K10 = invK_T[3], K11 = invK_T[4], K12 = invK_T[5];
    const float K20 = invK_T[6], K21 = invK_T[7], K22 = invK_T[8];

    float Y[9][3];
    float m0 = 0.f, m1 = 0.f, m2 = 0.f;
    int idx = 0;
#pragma unroll
    for (int dr = -1; dr <= 1; ++dr) {
#pragma unroll
        for (int dc = -1; dc <= 1; ++dc) {
            const int rr = r + dr, cc = c + dc;
            float x0 = 0.f, x1 = 0.f, x2 = 0.f;
            if (rr >= 0 && rr < H && cc >= 0 && cc < W) {
                const float d  = depth[rr * W + cc];
                const float px = ((float)cc + 0.5f) * d;
                const float py = ((float)rr + 0.5f) * d;
                x0 = px * K00 + py * K10 + d * K20;
                x1 = px * K01 + py * K11 + d * K21;
                x2 = px * K02 + py * K12 + d * K22;
            }
            Y[idx][0] = x0; Y[idx][1] = x1; Y[idx][2] = x2;
            m0 += x0; m1 += x1; m2 += x2;
            ++idx;
        }
    }
    m0 *= (1.f / 9.f); m1 *= (1.f / 9.f); m2 *= (1.f / 9.f);

    float C00 = 0.f, C01 = 0.f, C02 = 0.f, C11 = 0.f, C12 = 0.f, C22 = 0.f;
#pragma unroll
    for (int i = 0; i < 9; ++i) {
        const float dx = Y[i][0] - m0, dy = Y[i][1] - m1, dz = Y[i][2] - m2;
        C00 += dx * dx; C01 += dx * dy; C02 += dx * dz;
        C11 += dy * dy; C12 += dy * dz; C22 += dz * dz;
    }

    // smallest eigenvalue (trigonometric closed form for symmetric 3x3)
    const float q  = (C00 + C11 + C22) * (1.f / 3.f);
    const float p1 = C01 * C01 + C02 * C02 + C12 * C12;
    const float a00 = C00 - q, a11 = C11 - q, a22 = C22 - q;
    const float p2 = a00 * a00 + a11 * a11 + a22 * a22 + 2.f * p1;
    const float p  = sqrtf(p2 * (1.f / 6.f));

    float nx = 0.f, ny = 0.f, nz = 1.f;
    if (p > 1e-20f) {
        const float ip = 1.f / p;
        const float b00 = a00 * ip, b01 = C01 * ip, b02 = C02 * ip;
        const float b11 = a11 * ip, b12 = C12 * ip, b22 = a22 * ip;
        float detB = b00 * (b11 * b22 - b12 * b12)
                   - b01 * (b01 * b22 - b12 * b02)
                   + b02 * (b01 * b12 - b11 * b02);
        float rr2 = fminf(1.f, fmaxf(-1.f, 0.5f * detB));
        const float phi  = acosf(rr2) * (1.f / 3.f);
        const float lmin = q + 2.f * p * cosf(phi + 2.0943951023931953f); // +2pi/3

        const float A00 = C00 - lmin, A11 = C11 - lmin, A22 = C22 - lmin;
        // rows of (C - lmin I)
        const float c01x = C01 * C12 - C02 * A11, c01y = C02 * C01 - A00 * C12, c01z = A00 * A11 - C01 * C01;
        const float c02x = C01 * A22 - C02 * C12, c02y = C02 * C02 - A00 * A22, c02z = A00 * C12 - C01 * C02;
        const float c12x = A11 * A22 - C12 * C12, c12y = C12 * C02 - C01 * A22, c12z = C01 * C12 - A11 * C02;
        const float n01 = c01x * c01x + c01y * c01y + c01z * c01z;
        const float n02 = c02x * c02x + c02y * c02y + c02z * c02z;
        const float n12 = c12x * c12x + c12y * c12y + c12z * c12z;
        if (n01 >= n02 && n01 >= n12)      { nx = c01x; ny = c01y; nz = c01z; }
        else if (n02 >= n12)               { nx = c02x; ny = c02y; nz = c02z; }
        else                               { nx = c12x; ny = c12y; nz = c12z; }
        const float nrm = sqrtf(nx * nx + ny * ny + nz * nz);
        if (nrm > 1e-20f) { const float inv = 1.f / nrm; nx *= inv; ny *= inv; nz *= inv; }
        else              { nx = 0.f; ny = 0.f; nz = 1.f; }
    }
    if (nz > 0.f) { nx = -nx; ny = -ny; nz = -nz; }
    out[(size_t)P + (size_t)pix * 3 + 0] = nx;
    out[(size_t)P + (size_t)pix * 3 + 1] = ny;
    out[(size_t)P + (size_t)pix * 3 + 2] = nz;
}

// ---- launch ----------------------------------------------------------------

extern "C" void kernel_launch(void* const* d_in, const int* in_sizes, int n_in,
                              void* d_out, int out_size, void* d_ws, size_t ws_size,
                              hipStream_t stream) {
    const float* nxy    = (const float*)d_in[0];
    const float* sxy    = (const float*)d_in[1];
    // d_in[2] = xy (recomputed on the fly in normal_kernel)
    const float* mv     = (const float*)d_in[3];
    const float* invK_T = (const float*)d_in[4];
    const float* tab    = (const float*)d_in[5];
    const float* dW0    = (const float*)d_in[6];
    const float* db0    = (const float*)d_in[7];
    const float* dW1    = (const float*)d_in[8];
    const float* db1    = (const float*)d_in[9];
    const float* sW0    = (const float*)d_in[10];
    const float* sb0    = (const float*)d_in[11];
    const float* sW1    = (const float*)d_in[12];
    const float* sb1    = (const float*)d_in[13];
    const float* sW2    = (const float*)d_in[14];
    const float* sb2    = (const float*)d_in[15];

    const int P = in_sizes[0] / 2;
    const int W = (int)(sqrt((double)P) + 0.5);
    const int H = P / W;

    _Float16* wgt = (_Float16*)d_ws;
    float* out = (float*)d_out;

    hipLaunchKernelGGL(prep_weights, dim3((WGT_TOTAL + 255) / 256), dim3(256), 0, stream,
                       dW0, dW1, sW0, sW1, sW2, wgt);
    hipLaunchKernelGGL(fused_mlp, dim3(P / 64), dim3(128), 0, stream,
                       nxy, sxy, mv, tab, wgt, db0, db1, sb0, sb1, sb2, out, P);
    hipLaunchKernelGGL(normal_kernel, dim3((P + 255) / 256), dim3(256), 0, stream,
                       invK_T, out, out, P, W, H);
}
